// NoLayer_90005334655023
// MI455X (gfx1250) — matrix operand order repaired and checked by reading, not compiled, not run on hardware
//
#include <hip/hip_runtime.h>

// Problem constants (from reference)
#define N_IN   65536
#define N_OUT  16384
#define KNB    16          // neighbors K
#define N_DIST 4
#define N_PHI  8
#define MM     32          // N_DIST*N_PHI
#define FF     64          // feature dim
#define MF     2048        // MM*FF  (GEMM reduction length)
#define OUTC   256         // output channels

#define TILE_N    32       // n-rows per workgroup (2 WMMA row tiles)
#define AGG_PITCH 2072     // 2048 + 24 bf16 pad -> conflict-free b128 A loads

typedef __attribute__((ext_vector_type(16))) __bf16       v16bf;
typedef __attribute__((ext_vector_type(8)))  float        v8f;
typedef __attribute__((ext_vector_type(4)))  unsigned int u32x4;

static __device__ __forceinline__ unsigned short f2bf(float f) {
  unsigned int u = __builtin_bit_cast(unsigned int, f);
  u += 0x7fffu + ((u >> 16) & 1u);          // round-to-nearest-even
  return (unsigned short)(u >> 16);
}

static __device__ __forceinline__ v16bf pack16(u32x4 lo, u32x4 hi) {
  union { u32x4 q[2]; v16bf v; } u;
  u.q[0] = lo; u.q[1] = hi;
  return u.v;
}

struct Frags {
  u32x4 a0lo, a0hi, a1lo, a1hi;   // two A row-tiles (LDS)
  u32x4 b0lo, b0hi, b1lo, b1hi;   // two B col-tiles (global)
};

static __device__ __forceinline__ Frags
load_frags(const unsigned short* a0, const unsigned short* a1,
           const unsigned short* b0, const unsigned short* b1, int ks) {
  Frags f;
  const unsigned short* p;
  p = a0 + ks * 32; f.a0lo = *(const u32x4*)p; f.a0hi = *(const u32x4*)(p + 8);
  p = a1 + ks * 32; f.a1lo = *(const u32x4*)p; f.a1hi = *(const u32x4*)(p + 8);
  p = b0 + ks * 32; f.b0lo = *(const u32x4*)p; f.b0hi = *(const u32x4*)(p + 8);
  p = b1 + ks * 32; f.b1lo = *(const u32x4*)p; f.b1hi = *(const u32x4*)(p + 8);
  return f;
}

// ---------------------------------------------------------------------------
// Prep: W_out (2048 x 256, f32, row-major [k][col]) -> Wt bf16 [col][k]
// ---------------------------------------------------------------------------
__global__ __launch_bounds__(256)
void wout_bf16_transpose(const float* __restrict__ W, unsigned short* __restrict__ Wt) {
  int t   = blockIdx.x * 256 + threadIdx.x;   // 0 .. 2048*256-1
  int col = t & (OUTC - 1);
  int k   = t >> 8;
  Wt[(size_t)col * MF + k] = f2bf(W[(size_t)k * OUTC + col]);
}

// ---------------------------------------------------------------------------
// Fused: weights -> normalize -> gather -> agg (VALU) -> GEMM (WMMA bf16)
// One workgroup = 32 output rows; 8 waves, each owns 32 output columns and
// both 16-row tiles (4 accumulators; A shared across col-tiles, B across
// row-tiles).
// ---------------------------------------------------------------------------
__global__ __launch_bounds__(256)
void fused_neighborhood_gemm(const float* __restrict__ x,
                             const float* __restrict__ d_dists,
                             const float* __restrict__ d_phi,
                             const float* __restrict__ dists,
                             const float* __restrict__ sigma_p,
                             const float* __restrict__ kappa_p,
                             const float* __restrict__ phi,
                             const unsigned short* __restrict__ Wt,
                             const int* __restrict__ nh_idx,
                             float* __restrict__ out) {
  extern __shared__ char smem[];
  int*            lds_idx = (int*)smem;                          //  2 KB (512)
  float*          lds_w   = (float*)(smem + 2048);               // 64 KB (32n x 16k x 32m)
  float*          lds_x   = (float*)(smem + 2048 + 65536);       // 64 KB (16n x 16k x 64f, reused)
  unsigned short* lds_agg = (unsigned short*)(smem + 2048 + 65536 + 65536); // 32 x AGG_PITCH

  const int tid = threadIdx.x;
  const int n0  = blockIdx.x * TILE_N;

  // ---- Phase A1: per-(n,k) raw weights + neighbor indices (2 pairs/thread) --
  {
    const float inv_sigma = 1.0f / sigma_p[0];
    const float kap       = kappa_p[0];
#pragma unroll
    for (int j2 = 0; j2 < 2; ++j2) {
      const int pair = tid + j2 * 256;          // 0..511 = (nl*16 + k)
      const int nl = pair >> 4, k = pair & 15;
      const int gi = (n0 + nl) * KNB + k;
      lds_idx[pair] = nh_idx[gi];

      const float dd = d_dists[gi];
      const float dp = d_phi[gi];

      float wd[N_DIST];
#pragma unroll
      for (int j = 0; j < N_DIST; ++j) {
        float u = (dd - dists[j]) * inv_sigma;
        wd[j] = __expf(-0.5f * u * u);
      }
      float wp[N_PHI];
#pragma unroll
      for (int p = 0; p < N_PHI; ++p)
        wp[p] = __expf(kap * __cosf(dp - phi[p]));

      float* wrow = lds_w + pair * MM;
#pragma unroll
      for (int j = 0; j < N_DIST; ++j)
#pragma unroll
        for (int p = 0; p < N_PHI; ++p)
          wrow[j * N_PHI + p] = wd[j] * wp[p];  // m = j*8 + p
    }
  }
  __syncthreads();

  // ---- Phase A2: normalize over K per (n, m) (4 pairs/thread) ----
#pragma unroll
  for (int pp = 0; pp < 4; ++pp) {
    int pair = tid + pp * 256;                  // 0..1023 = 32n * 32m
    int nl = pair >> 5, m = pair & 31;
    float s = 0.0f;
#pragma unroll
    for (int k = 0; k < KNB; ++k) s += lds_w[(nl * KNB + k) * MM + m];
    float inv = 1.0f / (s + 1e-9f);
#pragma unroll
    for (int k = 0; k < KNB; ++k) lds_w[(nl * KNB + k) * MM + m] *= inv;
  }

  // ---- Phases B+C in two 16-row halves (lds_x reused) ----
  for (int h = 0; h < 2; ++h) {
    // B: gather x rows into LDS (16 threads per 64-float row, float4)
    {
      const int rg = tid >> 4, l16 = tid & 15;
#pragma unroll
      for (int it = 0; it < 16; ++it) {
        int lp = it * 16 + rg;                  // local (n,k) pair 0..255
        long idx = (long)lds_idx[h * 256 + lp];
        ((float4*)(lds_x + lp * FF))[l16] = ((const float4*)(x + idx * FF))[l16];
      }
    }
    __syncthreads();

    // C: agg[n][m*64 + f] = sum_k w[n,k,m] * x_nb[n,k,f] -> packed bf16x2
    {
      const int fp = tid & 31;                  // feature pair: f = 2*fp, 2*fp+1
      const int nb = tid >> 5;                  // uniform within a wave
#pragma unroll
      for (int j = 0; j < 2; ++j) {
        const int nl = nb + j * 8;              // 0..15 within half
        const int gn = h * 16 + nl;             // 0..31 agg row
        float2 xv[KNB];
#pragma unroll
        for (int k = 0; k < KNB; ++k)
          xv[k] = *(const float2*)(lds_x + (nl * KNB + k) * FF + 2 * fp);
        const float* wb = lds_w + gn * KNB * MM;
        unsigned int* arow = (unsigned int*)(lds_agg + gn * AGG_PITCH) + fp;
#pragma unroll
        for (int m = 0; m < MM; ++m) {
          float a0 = 0.0f, a1 = 0.0f;
#pragma unroll
          for (int k = 0; k < KNB; ++k) {
            float wv_ = wb[k * MM + m];         // lane-uniform -> broadcast
            a0 += xv[k].x * wv_;
            a1 += xv[k].y * wv_;
          }
          arow[m * 32] = (unsigned int)f2bf(a0) | ((unsigned int)f2bf(a1) << 16);
        }
      }
    }
    __syncthreads();
  }

  // ---- Phase D: out_tile(32x256) = agg_bf16(32x2048) @ Wt_bf16(2048x256) ----
  {
    const int wv   = tid >> 5;                  // wave 0..7 -> cols [wv*32, +32)
    const int lane = tid & 31;
    const int col  = lane & 15;
    const int sel  = lane >> 4;                 // half-wave select

    // A (16x32 bf16): lanes<16 hold K0..15; lanes>=16 K16..31
    const unsigned short* a0 = lds_agg + col * AGG_PITCH + sel * 16;
    const unsigned short* a1 = a0 + 16 * AGG_PITCH;
    // B (32x16 bf16): lanes<16 hold K0..15; lanes>=16 K16..31 (Wt is [col][k])
    const unsigned short* b0 = Wt + (size_t)(wv * 32 + col) * MF + sel * 16;
    const unsigned short* b1 = b0 + (size_t)16 * MF;

    v8f c00 = {0, 0, 0, 0, 0, 0, 0, 0};
    v8f c01 = {0, 0, 0, 0, 0, 0, 0, 0};
    v8f c10 = {0, 0, 0, 0, 0, 0, 0, 0};
    v8f c11 = {0, 0, 0, 0, 0, 0, 0, 0};

    Frags f = load_frags(a0, a1, b0, b1, 0);
    for (int ks = 0; ks < MF / 32 - 1; ++ks) {
      Frags fn = load_frags(a0, a1, b0, b1, ks + 1);     // overlap with WMMA
      __builtin_prefetch(b0 + (ks + 8) * 32, 0, 3);      // speculative, no guard
      __builtin_prefetch(b1 + (ks + 8) * 32, 0, 3);
      v16bf A0 = pack16(f.a0lo, f.a0hi), A1 = pack16(f.a1lo, f.a1hi);
      v16bf B0 = pack16(f.b0lo, f.b0hi), B1 = pack16(f.b1lo, f.b1hi);
      c00 = __builtin_amdgcn_wmma_f32_16x16x32_bf16(false, A0, false, B0, (short)0, c00, false, false);
      c01 = __builtin_amdgcn_wmma_f32_16x16x32_bf16(false, A0, false, B1, (short)0, c01, false, false);
      c10 = __builtin_amdgcn_wmma_f32_16x16x32_bf16(false, A1, false, B0, (short)0, c10, false, false);
      c11 = __builtin_amdgcn_wmma_f32_16x16x32_bf16(false, A1, false, B1, (short)0, c11, false, false);
      f = fn;
    }
    {
      v16bf A0 = pack16(f.a0lo, f.a0hi), A1 = pack16(f.a1lo, f.a1hi);
      v16bf B0 = pack16(f.b0lo, f.b0hi), B1 = pack16(f.b1lo, f.b1hi);
      c00 = __builtin_amdgcn_wmma_f32_16x16x32_bf16(false, A0, false, B0, (short)0, c00, false, false);
      c01 = __builtin_amdgcn_wmma_f32_16x16x32_bf16(false, A0, false, B1, (short)0, c01, false, false);
      c10 = __builtin_amdgcn_wmma_f32_16x16x32_bf16(false, A1, false, B0, (short)0, c10, false, false);
      c11 = __builtin_amdgcn_wmma_f32_16x16x32_bf16(false, A1, false, B1, (short)0, c11, false, false);
    }

    // C/D layout: vgpr i -> row i + 8*sel, col = lane&15
    float* o0 = out + (size_t)(n0 + sel * 8) * OUTC + wv * 32 + col;
    float* o1 = out + (size_t)(n0 + 16 + sel * 8) * OUTC + wv * 32 + col;
#pragma unroll
    for (int i = 0; i < 8; ++i) {
      o0[(size_t)i * OUTC]      = c00[i];
      o0[(size_t)i * OUTC + 16] = c01[i];
      o1[(size_t)i * OUTC]      = c10[i];
      o1[(size_t)i * OUTC + 16] = c11[i];
    }
  }
}

// ---------------------------------------------------------------------------
extern "C" void kernel_launch(void* const* d_in, const int* in_sizes, int n_in,
                              void* d_out, int out_size, void* d_ws, size_t ws_size,
                              hipStream_t stream) {
  const float* x       = (const float*)d_in[0];
  const float* d_dists = (const float*)d_in[1];
  const float* d_phi   = (const float*)d_in[2];
  const float* dists   = (const float*)d_in[3];
  const float* sigma   = (const float*)d_in[4];
  const float* kappa   = (const float*)d_in[5];
  const float* phi     = (const float*)d_in[6];
  const float* W_out   = (const float*)d_in[7];
  const int*   nh_idx  = (const int*)d_in[8];
  float*       out     = (float*)d_out;

  unsigned short* Wt = (unsigned short*)d_ws;  // 2048*256*2 = 1 MB scratch

  wout_bf16_transpose<<<(MF * OUTC) / 256, 256, 0, stream>>>(W_out, Wt);

  const size_t smem_bytes =
      2048 + 65536 + 65536 + (size_t)TILE_N * AGG_PITCH * 2;  // 265,728 B
  fused_neighborhood_gemm<<<N_OUT / TILE_N, 256, smem_bytes, stream>>>(
      x, d_dists, d_phi, dists, sigma, kappa, phi, Wt, nh_idx, out);
}
